// AutoRegRNN_63977832841338
// MI455X (gfx1250) — matrix operand order repaired and checked by reading, not compiled, not run on hardware
//
#include <hip/hip_runtime.h>

#define B_      512
#define T_      1024
#define F_      64
#define UNITS_  512
#define OUT_    64
#define STEPS_  128
#define ZCOLS   2048          // 4 * UNITS_
#define KTOT    576           // F_ + UNITS_
#define KC      32            // bf16 WMMA K per step
#define NKC     (KTOT / KC)   // 18
#define BM      128           // batch rows per workgroup
#define NWAVES  8

#if __has_builtin(__builtin_amdgcn_global_load_async_to_lds_b128)
#define HAS_ASYNC_LDS 1
#else
#define HAS_ASYNC_LDS 0
#endif

typedef __attribute__((ext_vector_type(16))) __bf16 v16bf;
typedef __attribute__((ext_vector_type(8)))  float  v8f;

// Async global->LDS builtin pointer types (from hipcc diagnostic:
// param 0 prints as 'vector_size(16) int __device__ *', i.e. v4i in AS(1))
typedef __attribute__((__vector_size__(16))) int v4i;
typedef __attribute__((address_space(1))) v4i* gv4i_p;
typedef __attribute__((address_space(3))) v4i* lv4i_p;

union AV { v16bf v; uint4 q[2]; unsigned short s[16]; };

__device__ __forceinline__ unsigned short f2bf(float f) {
    union { float f; unsigned u; } v; v.f = f;
    unsigned r = v.u + 0x7FFFu + ((v.u >> 16) & 1u);   // round-to-nearest-even
    return (unsigned short)(r >> 16);
}

__device__ __forceinline__ float sigm(float x) {
    return 1.0f / (1.0f + __expf(-x));
}

__device__ __forceinline__ void wait_async_zero() {
#if HAS_ASYNC_LDS
#if __has_builtin(__builtin_amdgcn_s_wait_asynccnt)
    __builtin_amdgcn_s_wait_asynccnt(0);
#else
    asm volatile("s_wait_asynccnt 0" ::: "memory");
#endif
#endif
}

// ---------------------------------------------------------------------------
// Weight packing:  WuT[col][k] bf16, col in [0,2048), k in [0,576)
//   k <  64 : W[k][col]       (input kernel,  F x 4U, row-major f32)
//   k >= 64 : U[k-64][col]    (recurrent,     U x 4U, row-major f32)
// ---------------------------------------------------------------------------
__global__ void pack_wu_kernel(const float* __restrict__ W,
                               const float* __restrict__ U,
                               unsigned short* __restrict__ WuT) {
    int idx = blockIdx.x * 256 + threadIdx.x;
    if (idx >= ZCOLS * KTOT) return;
    int col = idx / KTOT;
    int k   = idx - col * KTOT;
    float v = (k < F_) ? W[(size_t)k * ZCOLS + col]
                       : U[(size_t)(k - F_) * ZCOLS + col];
    WuT[idx] = f2bf(v);
}

// WdT[n][k] bf16 from Wd[k][n] (UNITS_ x OUT_ f32)
__global__ void pack_wd_kernel(const float* __restrict__ Wd,
                               unsigned short* __restrict__ WdT) {
    int idx = blockIdx.x * 256 + threadIdx.x;
    if (idx >= OUT_ * UNITS_) return;
    int n = idx / UNITS_;
    int k = idx - n * UNITS_;
    WdT[idx] = f2bf(Wd[(size_t)k * OUT_ + n]);
}

__global__ void init_state_kernel(float* __restrict__ c,
                                  unsigned short* __restrict__ h0) {
    int idx = blockIdx.x * 256 + threadIdx.x;
    if (idx >= B_ * UNITS_) return;
    c[idx]  = 0.0f;
    h0[idx] = 0;   // bf16 zero
}

// ---------------------------------------------------------------------------
// Fused LSTM step:
//   z[b, g*512+u] = x_t[b,:] @ W + h[b,:] @ U + bias   (bf16 WMMA, f32 accum)
//   i,f,g,o gates -> c_new, h_new fused in-register.
// Grid: (32 unit-tiles, 4 row-tiles), block = 256 (8 waves).
// Wave w: rows [row0 + 16w, +16), all 4 gates of the WG's 16-unit tile.
// h is ping-ponged across launches (h_read / h_write) so no cross-WG race.
// B tile is staged via async global->LDS copies (ASYNCcnt) when available.
// ---------------------------------------------------------------------------
__global__ __launch_bounds__(256)
void lstm_step_kernel(const float*          __restrict__ xin,     // [B_][x_stride] f32
                      int                               x_stride,
                      const unsigned short* __restrict__ WuT,     // [2048][576] bf16
                      const float*          __restrict__ bias,    // [2048]
                      const unsigned short* __restrict__ h_read,  // [B_][512] bf16
                      unsigned short*       __restrict__ h_write, // [B_][512] bf16
                      float*                __restrict__ c)       // [B_][512] f32
{
    __shared__ unsigned short Blds[2][64][KC];   // 8 KB, double-buffered B tile

    const int tid  = threadIdx.x;
    const int lane = tid & 31;
    const int wave = tid >> 5;
    const int u0   = blockIdx.x * 16;     // unit tile
    const int row0 = blockIdx.y * BM;     // batch rows of this WG
    const int wrow = row0 + wave * 16;    // this wave's 16 batch rows

    // --- cooperative B-tile loader: 64 z-cols (4 gates x 16 units) x 32 K ---
    const int bg   = tid >> 6;            // gate 0..3
    const int bi   = tid & 63;
    const int bcol = bi >> 2;             // unit 0..15 within tile
    const int bk8  = (bi & 3) * 8;        // K sub-offset {0,8,16,24}
    const unsigned short* bsrc =
        WuT + (size_t)(bg * UNITS_ + u0 + bcol) * KTOT + bk8;
    unsigned short* bdst0 = &Blds[0][bg * 16 + bcol][bk8];
    unsigned short* bdst1 = &Blds[1][bg * 16 + bcol][bk8];

    auto loadB = [&](int kc, int buf) {
        unsigned short* d = buf ? bdst1 : bdst0;
#if HAS_ASYNC_LDS
        // CDNA5 async copy: global -> LDS, no VGPR staging, tracked by ASYNCcnt
        __builtin_amdgcn_global_load_async_to_lds_b128(
            (gv4i_p)(bsrc + kc * KC),
            (lv4i_p)d,
            /*offset=*/0, /*cpol=*/0);
#else
        uint4 v = *(const uint4*)(bsrc + kc * KC);   // 8 bf16, 16B
        *(uint4*)d = v;                              // ds_store_b128
#endif
    };

    // --- per-lane WMMA operand geometry ---
    const int am   = wrow + (lane & 15);          // A row (batch)
    const int ak0  = (lane < 16) ? 0 : 8;         // A K grouping {0-7,16-23}/{8-15,24-31}
    const int bn   = lane & 15;                   // B column within 16
    const int bk0  = (lane < 16) ? 0 : 8;         // B K grouping {0-7,16-23}/{8-15,24-31}

    v8f acc[4];
    #pragma unroll
    for (int g = 0; g < 4; ++g) acc[g] = (v8f){0,0,0,0,0,0,0,0};

    loadB(0, 0);
    wait_async_zero();     // async LDS writes must complete before the barrier
    __syncthreads();

    for (int kc = 0; kc < NKC; ++kc) {
        const int cur = kc & 1;
        if (kc + 1 < NKC) loadB(kc + 1, cur ^ 1);   // overlaps with compute below

        // ---- A operand: K<64 from x_t (f32 -> bf16), K>=64 from h (bf16) ----
        AV a;
        if (kc < 2) {
            const float* s = xin + (size_t)am * x_stride + kc * KC;
            #pragma unroll
            for (int j = 0; j < 8; ++j) {
                a.s[j]     = f2bf(s[ak0 + j]);
                a.s[8 + j] = f2bf(s[ak0 + 16 + j]);
            }
        } else {
            const unsigned short* s = h_read + (size_t)am * UNITS_ + (kc * KC - F_);
            a.q[0] = *(const uint4*)(s + ak0);
            a.q[1] = *(const uint4*)(s + ak0 + 16);
        }

        // ---- 4 gate WMMAs against shared B tile ----
        #pragma unroll
        for (int g = 0; g < 4; ++g) {
            AV bb;
            const unsigned short* bp = &Blds[cur][g * 16 + bn][bk0];
            bb.q[0] = *(const uint4*)bp;        // ds_load_b128
            bb.q[1] = *(const uint4*)(bp + 16);
            acc[g] = __builtin_amdgcn_wmma_f32_16x16x32_bf16(
                         false, a.v, false, bb.v, (short)0, acc[g], false, false);
        }

        wait_async_zero();   // next buffer's async copies done before barrier
        __syncthreads();
    }

    // ---- fused gate epilogue ----
    // C/D layout: VGPR r, lanes 0-15 -> (M=r, N=lane); lanes 16-31 -> (M=r+8, N=lane-16)
    const int eu    = u0 + (lane & 15);
    const int erow0 = wrow + ((lane < 16) ? 0 : 8);
    const float bI = bias[eu];
    const float bF = bias[eu + UNITS_];
    const float bG = bias[eu + 2 * UNITS_];
    const float bO = bias[eu + 3 * UNITS_];

    #pragma unroll
    for (int r = 0; r < 8; ++r) {
        const size_t idx = (size_t)(erow0 + r) * UNITS_ + eu;
        const float i = sigm(acc[0][r] + bI);
        const float f = sigm(acc[1][r] + bF);
        const float g = tanhf(acc[2][r] + bG);
        const float o = sigm(acc[3][r] + bO);
        const float cn = f * c[idx] + i * g;
        c[idx] = cn;
        h_write[idx] = f2bf(o * tanhf(cn));
    }
}

// ---------------------------------------------------------------------------
// Dense head: pred = h @ Wd + bd   (M=512, K=512, N=64), bf16 WMMA.
// WdT is pre-transposed -> all operand loads contiguous, no LDS needed.
// Writes fp32 pred to d_out[:, s, :] and to pbuf (next-step input).
// Grid: 16 blocks x 256 threads; each wave owns one 16x16 tile.
// ---------------------------------------------------------------------------
__global__ __launch_bounds__(256)
void dense_kernel(const unsigned short* __restrict__ h,    // [B_][512] bf16
                  const unsigned short* __restrict__ WdT,  // [64][512] bf16
                  const float*          __restrict__ bd,   // [64]
                  float*                __restrict__ out,  // d_out + s*OUT_
                  int                               out_row_stride, // STEPS_*OUT_
                  float*                __restrict__ pbuf) // [B_][64] f32
{
    const int tid  = threadIdx.x;
    const int lane = tid & 31;
    const int wave = tid >> 5;
    const int tile = blockIdx.x * NWAVES + wave;   // 0..127
    const int rowT = tile >> 2;                    // 0..31
    const int colT = tile & 3;                     // 0..3

    const int am  = rowT * 16 + (lane & 15);
    const int ak0 = (lane < 16) ? 0 : 8;
    const int bn  = colT * 16 + (lane & 15);
    const int bk0 = (lane < 16) ? 0 : 8;

    v8f acc = (v8f){0,0,0,0,0,0,0,0};

    for (int kc = 0; kc < UNITS_ / KC; ++kc) {     // 16 iterations
        AV a, bb;
        const unsigned short* as = h + (size_t)am * UNITS_ + kc * KC;
        a.q[0] = *(const uint4*)(as + ak0);
        a.q[1] = *(const uint4*)(as + ak0 + 16);
        const unsigned short* bs = WdT + (size_t)bn * UNITS_ + kc * KC;
        bb.q[0] = *(const uint4*)(bs + bk0);
        bb.q[1] = *(const uint4*)(bs + bk0 + 16);
        acc = __builtin_amdgcn_wmma_f32_16x16x32_bf16(
                  false, a.v, false, bb.v, (short)0, acc, false, false);
    }

    const int col   = colT * 16 + (lane & 15);
    const int erow0 = rowT * 16 + ((lane < 16) ? 0 : 8);
    const float bb0 = bd[col];
    #pragma unroll
    for (int r = 0; r < 8; ++r) {
        const int row = erow0 + r;
        const float p = acc[r] + bb0;
        out[(size_t)row * out_row_stride + col] = p;
        pbuf[(size_t)row * OUT_ + col] = p;
    }
}

// ---------------------------------------------------------------------------
extern "C" void kernel_launch(void* const* d_in, const int* in_sizes, int n_in,
                              void* d_out, int out_size, void* d_ws, size_t ws_size,
                              hipStream_t stream) {
    const float* x  = (const float*)d_in[0];   // [B,T,F]
    const float* W  = (const float*)d_in[1];   // [F,4U]
    const float* U  = (const float*)d_in[2];   // [U,4U]
    const float* b  = (const float*)d_in[3];   // [4U]
    const float* Wd = (const float*)d_in[4];   // [U,OUT]
    const float* bd = (const float*)d_in[5];   // [OUT]
    float* out = (float*)d_out;                // [B, STEPS, OUT]

    char* ws = (char*)d_ws;
    unsigned short* WuT = (unsigned short*)ws; ws += (size_t)ZCOLS * KTOT * 2;   // 2.25 MB
    unsigned short* WdT = (unsigned short*)ws; ws += (size_t)OUT_ * UNITS_ * 2;  // 64 KB
    unsigned short* h0  = (unsigned short*)ws; ws += (size_t)B_ * UNITS_ * 2;    // 512 KB
    unsigned short* h1  = (unsigned short*)ws; ws += (size_t)B_ * UNITS_ * 2;    // 512 KB
    float*          c   = (float*)ws;          ws += (size_t)B_ * UNITS_ * 4;    // 1 MB
    float*          pb  = (float*)ws;          ws += (size_t)B_ * OUT_ * 4;      // 128 KB

    // 1) pack weights to bf16 (K-major), zero states — every call (deterministic)
    pack_wu_kernel<<<(ZCOLS * KTOT + 255) / 256, 256, 0, stream>>>(W, U, WuT);
    pack_wd_kernel<<<(OUT_ * UNITS_ + 255) / 256, 256, 0, stream>>>(Wd, WdT);
    init_state_kernel<<<(B_ * UNITS_ + 255) / 256, 256, 0, stream>>>(c, h0);

    unsigned short* hb[2] = { h0, h1 };
    dim3 gridL(UNITS_ / 16, B_ / BM);   // 32 x 4 = 128 WGs

    // 2) warmup: 1024 fused LSTM steps over x (step t reads hb[t&1], writes hb[1-(t&1)])
    for (int t = 0; t < T_; ++t) {
        const int p = t & 1;
        lstm_step_kernel<<<gridL, 256, 0, stream>>>(
            x + (size_t)t * F_, T_ * F_, WuT, b, hb[p], hb[1 - p], c);
    }

    // latest h is in hb[0] after 1024 steps
    dense_kernel<<<16, 256, 0, stream>>>(hb[0], WdT, bd, out, STEPS_ * OUT_, pb);

    // 3) autoregressive forecast: prediction feeds back as fp32 input (stride OUT_)
    for (int s = 1; s < STEPS_; ++s) {
        const int g = T_ + (s - 1);
        const int p = g & 1;
        lstm_step_kernel<<<gridL, 256, 0, stream>>>(
            pb, OUT_, WuT, b, hb[p], hb[1 - p], c);
        dense_kernel<<<16, 256, 0, stream>>>(
            hb[1 - p], WdT, bd, out + (size_t)s * OUT_, STEPS_ * OUT_, pb);
    }
}
